// SummaRuNNer_24103356465414
// MI455X (gfx1250) — matrix operand order repaired and checked by reading, not compiled, not run on hardware
//
#include <hip/hip_runtime.h>
#include <hip/hip_bf16.h>
#include <math.h>

// ---- problem constants ----
#define S_N   128    // sentences
#define W_N   100    // words per sentence
#define E_N   100    // embedding dim
#define HW    200    // word LSTM hidden
#define G4    800    // 4*HW gates
#define KE    128    // E padded to 4 k-tiles of 32
#define KH    224    // HW padded to 7 k-tiles of 32
#define HS    200    // sentence LSTM hidden
#define DS_IN 400    // 2*HW sentence LSTM input
#define PD_N  50

// ---- workspace byte offsets (all multiples of 64) ----
#define OFF_SEQLEN 0                         // 128 * 4
#define OFF_XW     1024                      // 12800*128 f16 = 3,276,800
#define OFF_WIHT   (OFF_XW   + 3276800)      // 2*128*800 f16 = 409,600
#define OFF_WHHT   (OFF_WIHT + 409600)       // 2*224*800 f16 = 716,800
#define OFF_BIAS   (OFF_WHHT + 716800)       // 2*800 f32 = 6,400
#define OFF_POOL   (OFF_BIAS + 6400)         // 2*128*200 f32 = 204,800
#define OFF_SFEAT  (OFF_POOL + 409600)       // 128*400 f32 = 204,800
#define OFF_SO     (OFF_SFEAT+ 204800)       // 128*400 f32 = 204,800

typedef _Float16     half16 __attribute__((ext_vector_type(16)));
typedef float        float8 __attribute__((ext_vector_type(8)));
typedef unsigned int u32x4  __attribute__((ext_vector_type(4)));
typedef float        f32x4  __attribute__((ext_vector_type(4)));

union Frag { half16 h; u32x4 q[2]; };

__device__ __forceinline__ float sigm(float x) { return 1.0f / (1.0f + __expf(-x)); }

// ---------------------------------------------------------------------------
// Prep kernels
// ---------------------------------------------------------------------------
__global__ void k_seqlen(const int* __restrict__ x, int* __restrict__ seqlen) {
  int s = threadIdx.x;
  if (s < S_N) {
    int c = 0;
    for (int w = 0; w < W_N; ++w) c += (x[s * W_N + w] != 0) ? 1 : 0;
    seqlen[s] = c < 1 ? 1 : c;
  }
}

// Xw[w*128 + s][k] = f16(word_emb[x[s][w]][k]), zero-padded k in [100,128)
__global__ void k_gather(const int* __restrict__ x, const float* __restrict__ emb,
                         _Float16* __restrict__ Xw) {
  int idx = blockIdx.x * blockDim.x + threadIdx.x;
  if (idx >= 12800 * KE) return;
  int m = idx >> 7, k = idx & 127;
  int w = m >> 7, s = m & 127;
  float v = 0.0f;
  if (k < E_N) v = emb[(size_t)x[s * W_N + w] * E_N + k];
  Xw[idx] = (_Float16)v;
}

// WihT[dir][k][n'] gate-interleaved: n' = 4*hu + g, src row = g*200+hu
__global__ void k_wih(const float* __restrict__ W0, const float* __restrict__ W1,
                      _Float16* __restrict__ dst) {
  int idx = blockIdx.x * blockDim.x + threadIdx.x;
  if (idx >= 2 * KE * G4) return;
  int dir = idx / (KE * G4), r = idx % (KE * G4);
  int k = r / G4, n = r % G4;
  int hu = n >> 2, g = n & 3;
  const float* W = dir ? W1 : W0;
  float v = (k < E_N) ? W[(size_t)(g * HW + hu) * E_N + k] : 0.0f;
  dst[idx] = (_Float16)v;
}

__global__ void k_whh(const float* __restrict__ W0, const float* __restrict__ W1,
                      _Float16* __restrict__ dst) {
  int idx = blockIdx.x * blockDim.x + threadIdx.x;
  if (idx >= 2 * KH * G4) return;
  int dir = idx / (KH * G4), r = idx % (KH * G4);
  int k = r / G4, n = r % G4;
  int hu = n >> 2, g = n & 3;
  const float* W = dir ? W1 : W0;
  float v = (k < HW) ? W[(size_t)(g * HW + hu) * HW + k] : 0.0f;
  dst[idx] = (_Float16)v;
}

__global__ void k_bias(const float* __restrict__ bi0, const float* __restrict__ bh0,
                       const float* __restrict__ bi1, const float* __restrict__ bh1,
                       float* __restrict__ dst) {
  int idx = blockIdx.x * blockDim.x + threadIdx.x;
  if (idx >= 2 * G4) return;
  int dir = idx / G4, n = idx % G4;
  int hu = n >> 2, g = n & 3;
  const float* bi = dir ? bi1 : bi0;
  const float* bh = dir ? bh1 : bh0;
  dst[idx] = bi[g * HW + hu] + bh[g * HW + hu];
}

// ---------------------------------------------------------------------------
// Word-level BiLSTM, WMMA f16 x f16 -> f32.  16 blocks: dir(2) x m-tile(8).
// Per step: gates[16x800] = bias + Xw(t)[16x128] @ WihT + h[16x224] @ WhhT
// ---------------------------------------------------------------------------
__global__ __launch_bounds__(256) void k_wordlstm(
    const _Float16* __restrict__ Xw, const _Float16* __restrict__ WihT,
    const _Float16* __restrict__ WhhT, const float* __restrict__ biasI,
    const int* __restrict__ seqlen, float* __restrict__ pool) {
  __shared__ __align__(16) _Float16 hbuf[2][16][KH];   // double-buffered h (f16)
  __shared__ __align__(16) float    stage[8][16][16];  // per-wave gate regroup tile
  __shared__ float cst[16][HW];                        // cell state
  __shared__ float sum[16][HW];                        // masked pool accumulator

  const int tid  = threadIdx.x;
  const int lane = tid & 31, wave = tid >> 5;
  const int dir  = blockIdx.x >> 3;
  const int mbase = (blockIdx.x & 7) * 16;

  for (int i = tid; i < 2 * 16 * KH; i += 256) (&hbuf[0][0][0])[i] = (_Float16)0.0f;
  for (int i = tid; i < 16 * HW; i += 256) { (&cst[0][0])[i] = 0.0f; (&sum[0][0])[i] = 0.0f; }
  __syncthreads();

  const int ml  = lane & 15;          // M row inside tile (also C-tile N index)
  const int lhi = lane >> 4;
  const int kf  = lhi * 8;            // A-frag K sub-offset per half-wave
  const int len = seqlen[mbase + ml]; // this lane's elements share one sentence

  const _Float16* WihTd = WihT + (size_t)dir * KE * G4;
  const _Float16* WhhTd = WhhT + (size_t)dir * KH * G4;
  const float*    biasd = biasI + dir * G4;

  for (int tp = 0; tp < W_N; ++tp) {
    const int tin = dir ? (W_N - 1 - tp) : tp;   // original word position
    const int cur = tp & 1, nxt = cur ^ 1;

    // A-fragments: token embeddings (global, f16, row-major [12800][128])
    Frag ax[4];
    const _Float16* xrow = Xw + (size_t)(tin * 128 + mbase + ml) * KE;
#pragma unroll
    for (int kt = 0; kt < 4; ++kt) {
      ax[kt].q[0] = *(const u32x4*)(xrow + kt * 32 + kf);
      ax[kt].q[1] = *(const u32x4*)(xrow + kt * 32 + kf + 16);
    }
    // A-fragments: previous hidden state (LDS)
    Frag ah[7];
    const _Float16* hrow = &hbuf[cur][ml][0];
#pragma unroll
    for (int kt = 0; kt < 7; ++kt) {
      ah[kt].q[0] = *(const u32x4*)(hrow + kt * 32 + kf);
      ah[kt].q[1] = *(const u32x4*)(hrow + kt * 32 + kf + 16);
    }

    for (int nt = wave; nt < 50; nt += 8) {       // wave-uniform tile ownership
      if (nt + 8 < 50) {                          // warm next weight tiles (L2)
        __builtin_prefetch(WihTd + (size_t)lane * G4 + (nt + 8) * 16, 0, 1);
        __builtin_prefetch(WhhTd + (size_t)lane * G4 + (nt + 8) * 16, 0, 1);
      }
      const float bv = biasd[nt * 16 + ml];
      float8 acc;
#pragma unroll
      for (int i = 0; i < 8; ++i) acc[i] = bv;

      const _Float16* bi = WihTd + (size_t)kf * G4 + nt * 16 + ml;  // B: lane = N col, elems = K
#pragma unroll
      for (int kt = 0; kt < 4; ++kt) {
        Frag b;
#pragma unroll
        for (int s = 0; s < 16; ++s) b.h[s] = bi[(size_t)(kt * 32 + (s < 8 ? s : s + 8)) * G4];
        acc = __builtin_amdgcn_wmma_f32_16x16x32_f16(false, ax[kt].h, false, b.h,
                                                     (short)0, acc, false, false);
      }
      const _Float16* bh = WhhTd + (size_t)kf * G4 + nt * 16 + ml;
#pragma unroll
      for (int kt = 0; kt < 7; ++kt) {
        Frag b;
#pragma unroll
        for (int s = 0; s < 16; ++s) b.h[s] = bh[(size_t)(kt * 32 + (s < 8 ? s : s + 8)) * G4];
        acc = __builtin_amdgcn_wmma_f32_16x16x32_f16(false, ah[kt].h, false, b.h,
                                                     (short)0, acc, false, false);
      }

      // stage C tile to LDS so each lane can gather its (m, hu)'s i/f/g/o
      asm volatile("" ::: "memory");
#pragma unroll
      for (int r = 0; r < 8; ++r) stage[wave][r + 8 * lhi][ml] = acc[r];
      asm volatile("s_wait_dscnt 0" ::: "memory");  // wave-local DS RAW fence

#pragma unroll
      for (int j = 0; j < 2; ++j) {
        const int e = lane + 32 * j;
        const int m = e & 15, hul = e >> 4;        // m == lane&15 for both j
        f32x4 gv = *(const f32x4*)&stage[wave][m][hul * 4];  // i, f, g, o
        const int hu = nt * 4 + hul;
        float cp = cst[m][hu];
        float cn = sigm(gv[1]) * cp + sigm(gv[0]) * tanhf(gv[2]);
        float hn = sigm(gv[3]) * tanhf(cn);
        cst[m][hu] = cn;
        hbuf[nxt][m][hu] = (_Float16)hn;
        if (tin < len) sum[m][hu] += hn;           // masked mean-pool (fused)
      }
      asm volatile("s_wait_dscnt 0" ::: "memory");
    }
    __syncthreads();   // all h(t+1) written before next step reads it
  }

  for (int i = tid; i < 16 * HW; i += 256) {
    int m = i / HW, hu = i % HW;
    pool[((size_t)dir * S_N + (mbase + m)) * HW + hu] = sum[m][hu];
  }
}

// sent_feat[s][col] = pool[dir][s][hu] / seq_len[s]
__global__ void k_pool(const float* __restrict__ pool, const int* __restrict__ seqlen,
                       float* __restrict__ sfeat) {
  int idx = blockIdx.x * blockDim.x + threadIdx.x;
  if (idx >= S_N * DS_IN) return;
  int s = idx / DS_IN, col = idx % DS_IN;
  int d = col >= HW ? 1 : 0, hu = col - d * HW;
  sfeat[idx] = pool[((size_t)d * S_N + s) * HW + hu] / (float)seqlen[s];
}

// ---------------------------------------------------------------------------
// Sentence-level BiLSTM (B=1, latency bound -> VALU f32, LDS state)
// ---------------------------------------------------------------------------
__global__ __launch_bounds__(256) void k_sentlstm(
    const float* __restrict__ sfeat,
    const float* __restrict__ Wih0, const float* __restrict__ Whh0,
    const float* __restrict__ bi0, const float* __restrict__ bh0,
    const float* __restrict__ Wih1, const float* __restrict__ Whh1,
    const float* __restrict__ bi1, const float* __restrict__ bh1,
    float* __restrict__ so) {
  __shared__ float xr[DS_IN], hs[HS], cs[HS], gt[G4];
  const int tid = threadIdx.x, dir = blockIdx.x;
  const float* Wih = dir ? Wih1 : Wih0;
  const float* Whh = dir ? Whh1 : Whh0;
  const float* bi  = dir ? bi1  : bi0;
  const float* bh  = dir ? bh1  : bh0;
  for (int i = tid; i < HS; i += 256) { hs[i] = 0.0f; cs[i] = 0.0f; }
  __syncthreads();
  for (int tp = 0; tp < S_N; ++tp) {
    const int tin = dir ? (S_N - 1 - tp) : tp;
    for (int k = tid; k < DS_IN; k += 256) xr[k] = sfeat[tin * DS_IN + k];
    __syncthreads();
    for (int n = tid; n < G4; n += 256) {   // natural gate order: i|f|g|o blocks
      float a = bi[n] + bh[n];
      const float* wr = Wih + (size_t)n * DS_IN;
      for (int k = 0; k < DS_IN; ++k) a += wr[k] * xr[k];
      const float* hr = Whh + (size_t)n * HS;
      for (int k = 0; k < HS; ++k) a += hr[k] * hs[k];
      gt[n] = a;
    }
    __syncthreads();
    if (tid < HS) {
      float c = sigm(gt[HS + tid]) * cs[tid] + sigm(gt[tid]) * tanhf(gt[2 * HS + tid]);
      float h = sigm(gt[3 * HS + tid]) * tanhf(c);
      cs[tid] = c; hs[tid] = h;
      so[(size_t)tin * DS_IN + dir * HS + tid] = h;
    }
    __syncthreads();
  }
}

// ---------------------------------------------------------------------------
// Head: doc vector, Hmat, content/salience/position, novelty scan
// ---------------------------------------------------------------------------
__global__ __launch_bounds__(256) void k_head(
    const float* __restrict__ so, const float* __restrict__ fc1w,
    const float* __restrict__ fc1b, const float* __restrict__ fc2w,
    const float* __restrict__ fc2b, const float* __restrict__ Wc,
    const float* __restrict__ Ws, const float* __restrict__ Wr,
    const float* __restrict__ Wp, const float* __restrict__ bconst,
    const float* __restrict__ pos_emb, float* __restrict__ out) {
  extern __shared__ float sm[];
  float* docfeat = sm;              // 400
  float* doc     = docfeat + 400;   // 100
  float* wsdoc   = doc + 100;       // 100
  float* base    = wsdoc + 100;     // 128
  float* state   = base + 128;      // 100
  float* red     = state + 100;     // 128
  float* Hm      = red + 128;       // 128*100
  float* Am      = Hm + 12800;      // 128*100
  const int tid = threadIdx.x;

  for (int col = tid; col < DS_IN; col += 256) {
    float a = 0.0f;
    for (int s = 0; s < S_N; ++s) a += so[(size_t)s * DS_IN + col];
    docfeat[col] = a * (1.0f / S_N);
  }
  __syncthreads();
  if (tid < 100) {
    float a = fc1b[tid];
    for (int k = 0; k < DS_IN; ++k) a += fc1w[(size_t)tid * DS_IN + k] * docfeat[k];
    doc[tid] = tanhf(a);
  }
  __syncthreads();
  if (tid < 100) {
    float a = 0.0f;
    for (int j = 0; j < 100; ++j) a += Ws[tid * 100 + j] * doc[j];
    wsdoc[tid] = a;
  }
  for (int idx = tid; idx < S_N * 100; idx += 256) {
    int s = idx / 100, j = idx % 100;
    float a = fc2b[j];
    for (int k = 0; k < DS_IN; ++k) a += fc2w[(size_t)j * DS_IN + k] * so[(size_t)s * DS_IN + k];
    Hm[idx] = tanhf(a);
  }
  __syncthreads();
  for (int idx = tid; idx < S_N * 100; idx += 256) {
    int s = idx / 100, j = idx % 100;
    float a = 0.0f;
    for (int k = 0; k < 100; ++k) a += Hm[s * 100 + k] * Wr[k * 100 + j];
    Am[idx] = a;
  }
  __syncthreads();
  if (tid < S_N) {
    float c = 0.0f, sal = 0.0f, p = 0.0f;
    for (int k = 0; k < 100; ++k) { c += Hm[tid * 100 + k] * Wc[k];
                                    sal += Hm[tid * 100 + k] * wsdoc[k]; }
    for (int d = 0; d < PD_N; ++d) p += pos_emb[tid * PD_N + d] * Wp[d];
    base[tid] = c + sal + p + bconst[0];
  }
  if (tid < 100) state[tid] = 0.0f;
  __syncthreads();

  for (int s = 0; s < S_N; ++s) {
    if (tid < 128) red[tid] = (tid < 100) ? Am[s * 100 + tid] * tanhf(state[tid]) : 0.0f;
    __syncthreads();
    for (int off = 64; off > 0; off >>= 1) {
      if (tid < off) red[tid] += red[tid + off];
      __syncthreads();
    }
    float prob = sigm(base[s] - red[0]);
    if (tid < 100) state[tid] += Hm[s * 100 + tid] * prob;
    if (tid == 0) out[s] = prob;
    __syncthreads();
  }
}

// ---------------------------------------------------------------------------
extern "C" void kernel_launch(void* const* d_in, const int* in_sizes, int n_in,
                              void* d_out, int out_size, void* d_ws, size_t ws_size,
                              hipStream_t stream) {
  (void)in_sizes; (void)n_in; (void)out_size; (void)ws_size;
  const int*   x        = (const int*)d_in[0];
  const float* word_emb = (const float*)d_in[1];
  const float* pos_emb  = (const float*)d_in[2];
  // word LSTM: fwd (Wih, Whh, bih, bhh), bwd (...)
  const float* WihWf = (const float*)d_in[3];
  const float* WhhWf = (const float*)d_in[4];
  const float* bihWf = (const float*)d_in[5];
  const float* bhhWf = (const float*)d_in[6];
  const float* WihWb = (const float*)d_in[7];
  const float* WhhWb = (const float*)d_in[8];
  const float* bihWb = (const float*)d_in[9];
  const float* bhhWb = (const float*)d_in[10];
  // sentence LSTM
  const float* WihSf = (const float*)d_in[11];
  const float* WhhSf = (const float*)d_in[12];
  const float* bihSf = (const float*)d_in[13];
  const float* bhhSf = (const float*)d_in[14];
  const float* WihSb = (const float*)d_in[15];
  const float* WhhSb = (const float*)d_in[16];
  const float* bihSb = (const float*)d_in[17];
  const float* bhhSb = (const float*)d_in[18];
  const float* fc1w = (const float*)d_in[19];
  const float* fc1b = (const float*)d_in[20];
  const float* fc2w = (const float*)d_in[21];
  const float* fc2b = (const float*)d_in[22];
  const float* Wc   = (const float*)d_in[23];
  const float* Ws   = (const float*)d_in[24];
  const float* Wr   = (const float*)d_in[25];
  const float* Wp   = (const float*)d_in[26];
  const float* bb   = (const float*)d_in[27];
  float* out = (float*)d_out;

  char* ws = (char*)d_ws;
  int*      seqlen = (int*)(ws + OFF_SEQLEN);
  _Float16* Xw     = (_Float16*)(ws + OFF_XW);
  _Float16* WihT   = (_Float16*)(ws + OFF_WIHT);
  _Float16* WhhT   = (_Float16*)(ws + OFF_WHHT);
  float*    biasI  = (float*)(ws + OFF_BIAS);
  float*    pool   = (float*)(ws + OFF_POOL);
  float*    sfeat  = (float*)(ws + OFF_SFEAT);
  float*    so     = (float*)(ws + OFF_SO);

  k_seqlen<<<1, 128, 0, stream>>>(x, seqlen);
  k_gather<<<(12800 * KE + 255) / 256, 256, 0, stream>>>(x, word_emb, Xw);
  k_wih<<<(2 * KE * G4 + 255) / 256, 256, 0, stream>>>(WihWf, WihWb, WihT);
  k_whh<<<(2 * KH * G4 + 255) / 256, 256, 0, stream>>>(WhhWf, WhhWb, WhhT);
  k_bias<<<(2 * G4 + 255) / 256, 256, 0, stream>>>(bihWf, bhhWf, bihWb, bhhWb, biasI);
  k_wordlstm<<<16, 256, 0, stream>>>(Xw, WihT, WhhT, biasI, seqlen, pool);
  k_pool<<<(S_N * DS_IN + 255) / 256, 256, 0, stream>>>(pool, seqlen, sfeat);
  k_sentlstm<<<2, 256, 0, stream>>>(sfeat, WihSf, WhhSf, bihSf, bhhSf,
                                    WihSb, WhhSb, bihSb, bhhSb, so);
  size_t head_smem = (size_t)(400 + 100 + 100 + 128 + 100 + 128 + 12800 + 12800) * 4;
  k_head<<<1, 256, head_smem, stream>>>(so, fc1w, fc1b, fc2w, fc2b, Wc, Ws, Wr,
                                        Wp, bb, pos_emb, out);
}
